// ImprovedNonlinearKalmanNet_70901320123080
// MI455X (gfx1250) — matrix operand order repaired and checked by reading, not compiled, not run on hardware
//
#include <hip/hip_runtime.h>
#include <cstdint>
#include <cstddef>

// ---------------------------------------------------------------------------
// ImprovedNonlinearKalmanNet for MI455X (gfx1250, wave32, WMMA).
// All large GEMMs run through v_wmma_f32_16x16x32_bf16 with fp32 accumulate;
// the Kalman state recurrence and small feature nets stay in exact fp32.
// ---------------------------------------------------------------------------

typedef __attribute__((ext_vector_type(16))) __bf16        v16bf;
typedef __attribute__((ext_vector_type(8)))  float         v8f;
typedef __attribute__((ext_vector_type(4)))  unsigned int  uv4;
typedef __attribute__((ext_vector_type(4)))  float         fv4;

union Frag { v16bf b; uv4 q[2]; };

__device__ __forceinline__ unsigned short f2bf(float x) {
  union { float f; unsigned u; } v; v.f = x;
  unsigned r = v.u + 0x7fffu + ((v.u >> 16) & 1u);   // round-to-nearest-even
  return (unsigned short)(r >> 16);
}

// ---------------------------------------------------------------------------
// C[256 x Nout] = act(A[256 x K] @ W^T + bias), W row-major [Nout x K] bf16.
// Grid: (Nout/256, 4); Block: 256 threads = 8 waves.
// Block tile 64x256; wave tile 32x64 = 2x4 WMMA 16x16 accumulators.
// Requires K % 32 == 0, Nout % 256 == 0 (true for every layer here).
// ---------------------------------------------------------------------------
__global__ __launch_bounds__(256) void gemm_bf16_wmma(
    const float* __restrict__ A,
    const unsigned short* __restrict__ Wb,
    const float* __restrict__ bias,
    float* __restrict__ C,
    int K, int Nout, int ldc, int act)
{
  __shared__ unsigned short lA[64][40];    // 64 rows x 32 k (pad to 40)
  __shared__ unsigned short lB[256][40];   // 256 cols x 32 k (pad to 40)

  const int tid  = threadIdx.x;
  const int m0   = blockIdx.y * 64;
  const int n0   = blockIdx.x * 256;
  const int wave = tid >> 5;
  const int lane = tid & 31;
  const int l16  = lane & 15;
  const int hi   = lane >> 4;
  const int wr   = wave >> 2;   // 0..1 : row group of 32
  const int wc   = wave & 3;    // 0..3 : col group of 64

  const v8f vzero = {0.f,0.f,0.f,0.f,0.f,0.f,0.f,0.f};
  v8f acc[2][4];
  #pragma unroll
  for (int i = 0; i < 2; ++i)
    #pragma unroll
    for (int j = 0; j < 4; ++j) acc[i][j] = vzero;

  const int arow = tid >> 2;
  const int aseg = tid & 3;

  for (int k0 = 0; k0 < K; k0 += 32) {
    // stage W tile: thread t loads 32 contiguous bf16 of row n0+t
    {
      const uv4* s = reinterpret_cast<const uv4*>(Wb + (size_t)(n0 + tid) * K + k0);
      uv4* d = reinterpret_cast<uv4*>(&lB[tid][0]);
      d[0] = s[0]; d[1] = s[1]; d[2] = s[2]; d[3] = s[3];
    }
    // stage A tile: 64x32 fp32 -> bf16 (8 values / thread)
    {
      const float* src = A + (size_t)(m0 + arow) * K + k0 + aseg * 8;
      fv4 f0 = reinterpret_cast<const fv4*>(src)[0];
      fv4 f1 = reinterpret_cast<const fv4*>(src)[1];
      unsigned short* d = &lA[arow][aseg * 8];
      d[0]=f2bf(f0.x); d[1]=f2bf(f0.y); d[2]=f2bf(f0.z); d[3]=f2bf(f0.w);
      d[4]=f2bf(f1.x); d[5]=f2bf(f1.y); d[6]=f2bf(f1.z); d[7]=f2bf(f1.w);
    }
    __syncthreads();

    // A fragment 16x32 (lane l16 = row M; lanes<16 hold K{0..7,16..23},
    // lanes>=16 hold K{8..15,24..31})  -- ISA 7.12.2 16-bit A layout.
    Frag fa[2], fb[4];
    #pragma unroll
    for (int mt = 0; mt < 2; ++mt) {
      const int r   = wr * 32 + mt * 16 + l16;
      const int klo = hi * 8;
      fa[mt].q[0] = *reinterpret_cast<const uv4*>(&lA[r][klo]);
      fa[mt].q[1] = *reinterpret_cast<const uv4*>(&lA[r][16 + klo]);
    }
    // B fragment 32x16 (lane l16 = col N; lanes<16 hold K{0..7,16..23},
    // lanes>=16 hold K{8..15,24..31}), column of B = row of W.
    #pragma unroll
    for (int nt = 0; nt < 4; ++nt) {
      const int r  = wc * 64 + nt * 16 + l16;
      const int ko = hi * 8;
      fb[nt].q[0] = *reinterpret_cast<const uv4*>(&lB[r][ko]);
      fb[nt].q[1] = *reinterpret_cast<const uv4*>(&lB[r][ko + 16]);
    }
    #pragma unroll
    for (int mt = 0; mt < 2; ++mt)
      #pragma unroll
      for (int nt = 0; nt < 4; ++nt)
        acc[mt][nt] = __builtin_amdgcn_wmma_f32_16x16x32_bf16(
            false, fa[mt].b, false, fb[nt].b, (short)0, acc[mt][nt], false, false);
    __syncthreads();
  }

  // C/D layout: VGPR v, lanes 0-15 -> M=v, lanes 16-31 -> M=v+8, N=l16.
  #pragma unroll
  for (int mt = 0; mt < 2; ++mt) {
    #pragma unroll
    for (int nt = 0; nt < 4; ++nt) {
      const int col = n0 + wc * 64 + nt * 16 + l16;
      const float bv = bias ? bias[col] : 0.f;
      #pragma unroll
      for (int v = 0; v < 8; ++v) {
        const int row = m0 + wr * 32 + mt * 16 + hi * 8 + v;
        float val = acc[mt][nt][v] + bv;
        if (act == 1) val = fmaxf(val, 0.f);
        C[(size_t)row * ldc + col] = val;
      }
    }
  }
}

// ---------------------------------------------------------------------------
// Weight prep: bf16-convert and (optionally) concat [W0 | W1] along K.
// ---------------------------------------------------------------------------
__global__ void prep_w_kernel(unsigned short* __restrict__ dst,
                              const float* __restrict__ s0, int K0,
                              const float* __restrict__ s1, int K1, int Nout)
{
  const int Kc = K0 + K1;
  const size_t total = (size_t)Nout * Kc;
  for (size_t idx = (size_t)blockIdx.x * blockDim.x + threadIdx.x; idx < total;
       idx += (size_t)gridDim.x * blockDim.x) {
    const int n = (int)(idx / Kc);
    const int k = (int)(idx - (size_t)n * Kc);
    const float v = (k < K0) ? s0[(size_t)n * K0 + k] : s1[(size_t)n * K1 + (k - K0)];
    dst[idx] = f2bf(v);
  }
}

__global__ void addvec_kernel(float* __restrict__ d, const float* __restrict__ a,
                              const float* __restrict__ b, int n)
{
  int i = blockIdx.x * blockDim.x + threadIdx.x;
  if (i < n) d[i] = a[i] + b[i];
}

__global__ void zero_kernel(float* __restrict__ p, int n)
{
  for (int i = blockIdx.x * blockDim.x + threadIdx.x; i < n; i += gridDim.x * blockDim.x)
    p[i] = 0.f;
}

__global__ void yprev0_kernel(const float* __restrict__ x0, float* __restrict__ yp)
{
  int i = blockIdx.x * blockDim.x + threadIdx.x;     // B*16
  if (i < 256 * 16) {
    int b = i >> 4, n = i & 15;
    yp[i] = tanhf(x0[b * 32 + n]);
  }
}

__global__ void copy_cols_kernel(float* __restrict__ dst, int dstStride, int dstOff,
                                 const float* __restrict__ src, int srcStride, int srcOff,
                                 int cols)
{
  int i = blockIdx.x * blockDim.x + threadIdx.x;
  int total = 256 * cols;
  if (i < total) {
    int b = i / cols, c = i - b * cols;
    dst[(size_t)b * dstStride + dstOff + c] = src[(size_t)b * srcStride + srcOff + c];
  }
}

// ---------------------------------------------------------------------------
// Per-step features: x_prior=f(x_post); y_pred=h(x_prior); 4 normalized
// feature diffs; fc5/fc6/fc7 relu matvecs; assemble inQ = [out5 | hQ_prev].
// One block (64 threads) per batch element.
// ---------------------------------------------------------------------------
__global__ __launch_bounds__(64) void features_kernel(
    const float* __restrict__ Yt,         // [B,16]
    const float* __restrict__ xPost,      // [B,32]
    const float* __restrict__ xPostPrev,  // [B,32]
    const float* __restrict__ xPriorPrev, // [B,32]
    const float* __restrict__ yPrev,      // [B,16]
    const float* __restrict__ hQ,         // [B,1024]
    const float* __restrict__ fc5w, const float* __restrict__ fc5b,
    const float* __restrict__ fc6w, const float* __restrict__ fc6b,
    const float* __restrict__ fc7w, const float* __restrict__ fc7b,
    float* __restrict__ xPrior,           // [B,32]
    float* __restrict__ yPred,            // [B,16]
    float* __restrict__ inQ,              // [B,1184]
    float* __restrict__ out6, float* __restrict__ out7)  // [B,160]
{
  const int b = blockIdx.x;
  const int t = threadIdx.x;
  __shared__ float s_xprior[32], s_fwev[32], s_fwup[32], s_in7[32];
  __shared__ float s_inv[4];

  if (t < 32) {
    float xp  = xPost[b * 32 + t];
    float xpr = xp + 0.1f * sinf(xp);
    s_xprior[t] = xpr;
    xPrior[b * 32 + t] = xpr;
    s_fwev[t] = xp - xPostPrev[b * 32 + t];
    s_fwup[t] = xp - xPriorPrev[b * 32 + t];
  }
  __syncthreads();
  if (t < 16) {
    float yp = tanhf(s_xprior[t]);
    yPred[b * 16 + t] = yp;
    float yv = Yt[b * 16 + t];
    s_in7[t]      = yv - yPrev[b * 16 + t];  // obs_diff raw
    s_in7[16 + t] = yv - yp;                 // obs_innov raw
  }
  __syncthreads();
  if (t < 4) {
    const float* src; int n;
    if      (t == 0) { src = s_fwup;     n = 32; }
    else if (t == 1) { src = s_fwev;     n = 32; }
    else if (t == 2) { src = s_in7;      n = 16; }
    else             { src = s_in7 + 16; n = 16; }
    float ss = 0.f;
    for (int i = 0; i < n; ++i) ss += src[i] * src[i];
    s_inv[t] = 1.0f / fmaxf(sqrtf(ss), 1e-12f);
  }
  __syncthreads();
  if (t < 32) { s_fwup[t] *= s_inv[0]; s_fwev[t] *= s_inv[1]; }
  if (t < 16) { s_in7[t]  *= s_inv[2]; s_in7[16 + t] *= s_inv[3]; }
  __syncthreads();

  for (int j = t; j < 160; j += 64) {
    float a5 = fc5b[j], a6 = fc6b[j], a7 = fc7b[j];
    #pragma unroll 8
    for (int k = 0; k < 32; ++k) {
      a5 += fc5w[j * 32 + k] * s_fwup[k];
      a6 += fc6w[j * 32 + k] * s_fwev[k];
      a7 += fc7w[j * 32 + k] * s_in7[k];
    }
    inQ[(size_t)b * 1184 + j] = fmaxf(a5, 0.f);
    out6[(size_t)b * 160 + j] = fmaxf(a6, 0.f);
    out7[(size_t)b * 160 + j] = fmaxf(a7, 0.f);
  }
  for (int i = t; i < 1024; i += 64)
    inQ[(size_t)b * 1184 + 160 + i] = hQ[(size_t)b * 1024 + i];
}

// ---------------------------------------------------------------------------
// LSTM gate fusion (torch order i,f,g,o); optionally scatter new h into a
// downstream concat buffer (saves a copy kernel).
// ---------------------------------------------------------------------------
__global__ void lstm_gate_kernel(const float* __restrict__ z, float* __restrict__ h,
                                 float* __restrict__ c, float* __restrict__ hDst,
                                 int dstStride, int dstOff, int H)
{
  int idx = blockIdx.x * blockDim.x + threadIdx.x;
  int total = 256 * H;
  if (idx >= total) return;
  int b = idx / H, j = idx - b * H;
  const float* zb = z + (size_t)b * 4 * H;
  float zi = zb[j], zf = zb[H + j], zg = zb[2 * H + j], zo = zb[3 * H + j];
  float ig = 1.f / (1.f + expf(-zi));
  float fg = 1.f / (1.f + expf(-zf));
  float og = 1.f / (1.f + expf(-zo));
  float cn = fg * c[idx] + ig * tanhf(zg);
  float hn = og * tanhf(cn);
  c[idx] = cn; h[idx] = hn;
  if (hDst) hDst[(size_t)b * dstStride + dstOff + j] = hn;
}

// ---------------------------------------------------------------------------
// KF update: x_post = x_prior + K @ (y - y_pred); K = kvec.reshape(B,32,16).
// Writes the output slice directly (which is also next step's x_post carry).
// ---------------------------------------------------------------------------
__global__ __launch_bounds__(32) void update_kernel(
    const float* __restrict__ xPrior, const float* __restrict__ kvec,
    const float* __restrict__ Yt, const float* __restrict__ yPred,
    float* __restrict__ xOut)
{
  const int b = blockIdx.x, m = threadIdx.x;
  __shared__ float dy[16];
  if (m < 16) dy[m] = Yt[b * 16 + m] - yPred[b * 16 + m];
  __syncthreads();
  float a = xPrior[b * 32 + m];
  const float* kr = kvec + (size_t)b * 512 + m * 16;
  #pragma unroll
  for (int n = 0; n < 16; ++n) a += kr[n] * dy[n];
  xOut[b * 32 + m] = a;
}

// ---------------------------------------------------------------------------
extern "C" void kernel_launch(void* const* d_in, const int* in_sizes, int n_in,
                              void* d_out, int out_size, void* d_ws, size_t ws_size,
                              hipStream_t stream)
{
  const float* Y       = (const float*)d_in[0];
  const float* x0      = (const float*)d_in[1];
  const float* Wq_ih   = (const float*)d_in[2];
  const float* Wq_hh   = (const float*)d_in[3];
  const float* bq_ih   = (const float*)d_in[4];
  const float* bq_hh   = (const float*)d_in[5];
  const float* Wsig_ih = (const float*)d_in[6];
  const float* Wsig_hh = (const float*)d_in[7];
  const float* bsig_ih = (const float*)d_in[8];
  const float* bsig_hh = (const float*)d_in[9];
  const float* Ws_ih   = (const float*)d_in[10];
  const float* Ws_hh   = (const float*)d_in[11];
  const float* bs_ih   = (const float*)d_in[12];
  const float* bs_hh   = (const float*)d_in[13];
  const float* fc1_w   = (const float*)d_in[14];
  const float* fc1_b   = (const float*)d_in[15];
  const float* fc2a_w  = (const float*)d_in[16];
  const float* fc2a_b  = (const float*)d_in[17];
  const float* fc2b_w  = (const float*)d_in[18];
  const float* fc2b_b  = (const float*)d_in[19];
  const float* fc5_w   = (const float*)d_in[20];
  const float* fc5_b   = (const float*)d_in[21];
  const float* fc6_w   = (const float*)d_in[22];
  const float* fc6_b   = (const float*)d_in[23];
  const float* fc7_w   = (const float*)d_in[24];
  const float* fc7_b   = (const float*)d_in[25];
  float* out = (float*)d_out;

  uint8_t* base = (uint8_t*)d_ws;
  size_t off = 0;
  auto alloc = [&](size_t bytes) -> void* {
    off = (off + 255) & ~(size_t)255;
    void* p = base + off;
    off += bytes;
    return p;
  };

  // bf16 weights (LSTMs: [W_ih | W_hh] concat along K)
  unsigned short* WqCat   = (unsigned short*)alloc((size_t)4096 * 1184 * 2);
  unsigned short* WsigCat = (unsigned short*)alloc((size_t)4096 * 2208 * 2);
  unsigned short* WsCat   = (unsigned short*)alloc((size_t)1024 * 672  * 2);
  unsigned short* Wfc1    = (unsigned short*)alloc((size_t)256  * 1024 * 2);
  unsigned short* Wfc2a   = (unsigned short*)alloc((size_t)2560 * 1280 * 2);
  unsigned short* Wfc2b   = (unsigned short*)alloc((size_t)512  * 2560 * 2);
  // combined LSTM biases
  float* bq   = (float*)alloc(4096 * 4);
  float* bsig = (float*)alloc(4096 * 4);
  float* bs   = (float*)alloc(1024 * 4);
  // recurrent state
  float* hQ   = (float*)alloc((size_t)256 * 1024 * 4);
  float* cQ   = (float*)alloc((size_t)256 * 1024 * 4);
  float* hSig = (float*)alloc((size_t)256 * 1024 * 4);
  float* cSig = (float*)alloc((size_t)256 * 1024 * 4);
  float* hS   = (float*)alloc((size_t)256 * 256 * 4);
  float* cS   = (float*)alloc((size_t)256 * 256 * 4);
  // activations
  float* inQ   = (float*)alloc((size_t)256 * 1184 * 4);
  float* zQ    = (float*)alloc((size_t)256 * 4096 * 4);
  float* inSig = (float*)alloc((size_t)256 * 2208 * 4);
  float* zSig  = (float*)alloc((size_t)256 * 4096 * 4);
  float* inS   = (float*)alloc((size_t)256 * 672 * 4);
  float* zS    = (float*)alloc((size_t)256 * 1024 * 4);
  float* inFc2 = (float*)alloc((size_t)256 * 1280 * 4);
  float* h2    = (float*)alloc((size_t)256 * 2560 * 4);
  float* kvec  = (float*)alloc((size_t)256 * 512 * 4);
  float* out6b = (float*)alloc((size_t)256 * 160 * 4);
  float* out7b = (float*)alloc((size_t)256 * 160 * 4);
  float* xPriorBuf0 = (float*)alloc(256 * 32 * 4);
  float* xPriorBuf1 = (float*)alloc(256 * 32 * 4);
  float* yPred  = (float*)alloc(256 * 16 * 4);
  float* yprev0 = (float*)alloc(256 * 16 * 4);
  float* xPriorBuf[2] = { xPriorBuf0, xPriorBuf1 };

  // ---- one-time prep (re-done every call: deterministic, no cached state)
  prep_w_kernel<<<2048, 256, 0, stream>>>(WqCat,   Wq_ih,   160,  Wq_hh,   1024, 4096);
  prep_w_kernel<<<2048, 256, 0, stream>>>(WsigCat, Wsig_ih, 1184, Wsig_hh, 1024, 4096);
  prep_w_kernel<<<1024, 256, 0, stream>>>(WsCat,   Ws_ih,   416,  Ws_hh,   256,  1024);
  prep_w_kernel<<<512,  256, 0, stream>>>(Wfc1,    fc1_w,   1024, nullptr, 0,    256);
  prep_w_kernel<<<2048, 256, 0, stream>>>(Wfc2a,   fc2a_w,  1280, nullptr, 0,    2560);
  prep_w_kernel<<<1024, 256, 0, stream>>>(Wfc2b,   fc2b_w,  2560, nullptr, 0,    512);
  addvec_kernel<<<16, 256, 0, stream>>>(bq,   bq_ih,   bq_hh,   4096);
  addvec_kernel<<<16, 256, 0, stream>>>(bsig, bsig_ih, bsig_hh, 4096);
  addvec_kernel<<<4,  256, 0, stream>>>(bs,   bs_ih,   bs_hh,   1024);
  zero_kernel<<<512, 256, 0, stream>>>(hQ,   256 * 1024);
  zero_kernel<<<512, 256, 0, stream>>>(cQ,   256 * 1024);
  zero_kernel<<<512, 256, 0, stream>>>(hSig, 256 * 1024);
  zero_kernel<<<512, 256, 0, stream>>>(cSig, 256 * 1024);
  zero_kernel<<<128, 256, 0, stream>>>(hS,   256 * 256);
  zero_kernel<<<128, 256, 0, stream>>>(cS,   256 * 256);
  yprev0_kernel<<<16, 256, 0, stream>>>(x0, yprev0);

  // ---- T sequential steps
  for (int t = 0; t < 8; ++t) {
    const float* Yt         = Y + (size_t)t * 256 * 16;
    const float* xPost      = (t == 0) ? x0 : out + (size_t)(t - 1) * 8192;
    const float* xPostPrev  = (t <= 1) ? x0 : out + (size_t)(t - 2) * 8192;
    const float* xPriorPrev = (t == 0) ? x0 : xPriorBuf[(t - 1) & 1];
    const float* yPrev      = (t == 0) ? yprev0 : Y + (size_t)(t - 1) * 256 * 16;
    float* xPrior = xPriorBuf[t & 1];

    features_kernel<<<256, 64, 0, stream>>>(Yt, xPost, xPostPrev, xPriorPrev, yPrev, hQ,
                                            fc5_w, fc5_b, fc6_w, fc6_b, fc7_w, fc7_b,
                                            xPrior, yPred, inQ, out6b, out7b);
    // LSTM Q: zQ = [out5|hQ] @ [Wq_ih|Wq_hh]^T + bq
    gemm_bf16_wmma<<<dim3(16, 4), 256, 0, stream>>>(inQ, WqCat, bq, zQ, 1184, 4096, 4096, 0);
    lstm_gate_kernel<<<1024, 256, 0, stream>>>(zQ, hQ, cQ, inSig, 2208, 0, 1024);
    copy_cols_kernel<<<160, 256, 0, stream>>>(inSig, 2208, 1024, out6b, 160, 0, 160);
    copy_cols_kernel<<<1024, 256, 0, stream>>>(inSig, 2208, 1184, hSig, 1024, 0, 1024);
    // LSTM Sigma
    gemm_bf16_wmma<<<dim3(16, 4), 256, 0, stream>>>(inSig, WsigCat, bsig, zSig, 2208, 4096, 4096, 0);
    lstm_gate_kernel<<<1024, 256, 0, stream>>>(zSig, hSig, cSig, inFc2, 1280, 0, 1024);
    // fc1 (relu) written straight into inS cols [0,256)
    gemm_bf16_wmma<<<dim3(1, 4), 256, 0, stream>>>(hSig, Wfc1, fc1_b, inS, 1024, 256, 672, 1);
    copy_cols_kernel<<<160, 256, 0, stream>>>(inS, 672, 256, out7b, 160, 0, 160);
    copy_cols_kernel<<<256, 256, 0, stream>>>(inS, 672, 416, hS, 256, 0, 256);
    // LSTM S
    gemm_bf16_wmma<<<dim3(4, 4), 256, 0, stream>>>(inS, WsCat, bs, zS, 672, 1024, 1024, 0);
    lstm_gate_kernel<<<256, 256, 0, stream>>>(zS, hS, cS, inFc2, 1280, 1024, 256);
    // fc2a (relu) -> fc2b
    gemm_bf16_wmma<<<dim3(10, 4), 256, 0, stream>>>(inFc2, Wfc2a, fc2a_b, h2, 1280, 2560, 2560, 1);
    gemm_bf16_wmma<<<dim3(2, 4), 256, 0, stream>>>(h2, Wfc2b, fc2b_b, kvec, 2560, 512, 512, 0);
    // KF update -> output slice t (doubles as next x_post carry)
    update_kernel<<<256, 32, 0, stream>>>(xPrior, kvec, Yt, yPred, out + (size_t)t * 8192);
  }
}
